// CASiameseTransformer_45689862094963
// MI455X (gfx1250) — matrix operand order repaired and checked by reading, not compiled, not run on hardware
//
#include <hip/hip_runtime.h>

// ---------------------------------------------------------------------------
// Types for CDNA5 WMMA (bf16 inputs, f32 accumulate)
// ---------------------------------------------------------------------------
typedef __attribute__((ext_vector_type(16))) __bf16 v16bf;
typedef __attribute__((ext_vector_type(8)))  float  v8f;

// ---------------------------------------------------------------------------
// Fast strided, batched GEMM: C = alpha*(A@B) (+bias) (+C) (relu optional)
//   A element (m,k): A[m*sAm + k]        (bf16, k contiguous)
//   B element (k,n): BT ? B[k*sBk + n]   (n contiguous)  : B[k + n*sBn] (k contiguous)
//   C element (m,n): C[m*sCm + n]        (f32)
// No bounds checks: requires M%64==0, K%32==0, N%TBN==0, and all A/B base
// offsets + strides 16B-aligned (true for every call site below).
// flags: bit0 = accumulate into C, bit1 = relu
// Instantiated only as <64,false>, <64,true>, <32,true>.
// ---------------------------------------------------------------------------
template <int TBN, bool BT>
__global__ __launch_bounds__(128)
void gemm_fast_kernel(const __bf16* __restrict__ A, const __bf16* __restrict__ B,
                      float* __restrict__ C, const float* __restrict__ bias,
                      int K, long sAm, long sBk, long sBn, long sCm,
                      long bA, long bB, long bC,
                      float alpha, int flags) {
  constexpr int BM = 64, BK = 32;
  constexpr int WC = (TBN == 64) ? 2 : 1;   // wave grid cols
  constexpr int TM = (TBN == 64) ? 32 : 16; // per-wave M
  constexpr int AI = TM / 16;               // A fragments per wave
  constexpr int BROWS = BT ? BK : TBN;
  constexpr int BCOLS = BT ? (TBN + 8) : (BK + 8);

  __shared__ __bf16 As[BM][BK + 8];
  __shared__ __bf16 Bs[BROWS][BCOLS];

  const int tid  = threadIdx.x;
  const int lane = tid & 31;                // wave32
  const int wave = tid >> 5;                // 0..3
  const int wm   = (wave / WC) * TM;
  const int wn   = (wave % WC) * 32;
  const int m0   = blockIdx.y * BM;
  const int n0   = blockIdx.x * TBN;
  A += (size_t)blockIdx.z * bA;
  B += (size_t)blockIdx.z * bB;
  C += (size_t)blockIdx.z * bC;

  const int mrow = lane & 15;
  const int hf   = lane >> 4;

  // ---- straight-line staging setup (2 x b128 per thread per matrix) ----
  const int sar = tid >> 2;                 // 0..31
  const int sak = (tid & 3) << 3;           // 0,8,16,24
  const __bf16* aG = A + (size_t)(m0 + sar) * sAm + sak;
  const size_t  aStep2 = (size_t)32 * sAm;  // second row chunk
  __bf16* aL0 = &As[sar][sak];
  __bf16* aL1 = &As[sar + 32][sak];

  const __bf16* bG;
  __bf16 *bL0, *bL1 = nullptr;
  size_t bStride2 = 0;                      // offset of second chunk
  long   bStepK;                            // advance per K-step
  if constexpr (BT) {
    constexpr int NW = TBN / 8;
    const int kk = tid / NW, nq = (tid % NW) << 3;
    bG = B + (size_t)kk * sBk + (n0 + nq);
    bL0 = &Bs[kk][nq];
    if constexpr (TBN == 64) { bL1 = &Bs[kk + 16][nq]; bStride2 = (size_t)16 * sBk; }
    bStepK = (long)BK * sBk;
  } else {
    bG = B + (size_t)(n0 + sar) * sBn + sak;
    bL0 = &Bs[sar][sak];
    bL1 = &Bs[sar + 32][sak];
    bStride2 = (size_t)32 * sBn;
    bStepK = BK;
  }

  v8f acc[AI][2] = {};

  for (int k0 = 0; k0 < K; k0 += BK) {
    *(uint4*)aL0 = *(const uint4*)aG;
    *(uint4*)aL1 = *(const uint4*)(aG + aStep2);
    *(uint4*)bL0 = *(const uint4*)bG;
    if constexpr (!BT || TBN == 64)
      *(uint4*)bL1 = *(const uint4*)(bG + bStride2);
    __syncthreads();

    v16bf aF[AI], bF[2];
#pragma unroll
    for (int t = 0; t < AI; ++t)
#pragma unroll
      for (int e = 0; e < 16; ++e) {
        // A 16x32 bf16 fragment: k = 16*hf + e (matches B)
        int ka = 16 * hf + e;
        aF[t][e] = As[wm + t * 16 + mrow][ka];
      }
#pragma unroll
    for (int t = 0; t < 2; ++t)
#pragma unroll
      for (int e = 0; e < 16; ++e) {
        // B 32x16 bf16 fragment: k = 16*hf + e, n = lane&15
        int kb = 16 * hf + e;
        if constexpr (BT) bF[t][e] = Bs[kb][wn + t * 16 + mrow];
        else              bF[t][e] = Bs[wn + t * 16 + mrow][kb];
      }
#pragma unroll
    for (int i = 0; i < AI; ++i)
#pragma unroll
      for (int j = 0; j < 2; ++j)
        acc[i][j] = __builtin_amdgcn_wmma_f32_16x16x32_bf16(
            false, aF[i], false, bF[j], (short)0, acc[i][j], false, false);
    __syncthreads();

    aG += BK;
    bG += bStepK;
  }

  // epilogue: C element (VGPR r): m = r + 8*hf, n = lane&15
#pragma unroll
  for (int i = 0; i < AI; ++i)
#pragma unroll
    for (int j = 0; j < 2; ++j) {
      int ng = n0 + wn + j * 16 + mrow;
      float bv = bias ? bias[ng] : 0.0f;
#pragma unroll
      for (int r = 0; r < 8; ++r) {
        int mg = m0 + wm + i * 16 + 8 * hf + r;
        float v = alpha * acc[i][j][r] + bv;
        size_t ci = (size_t)mg * sCm + ng;
        if (flags & 1) v += C[ci];
        if (flags & 2) v = fmaxf(v, 0.f);
        C[ci] = v;
      }
    }
}

// ---------------------------------------------------------------------------
// Elementwise / small kernels
// ---------------------------------------------------------------------------
__global__ void pack_bf16_kernel(const float* __restrict__ x, __bf16* __restrict__ y, long n) {
  long i = (long)blockIdx.x * 256 + threadIdx.x;
  if (i < n) y[i] = (__bf16)x[i];
}

__global__ void zero_kernel(float* p, long n) {
  long i = (long)blockIdx.x * 256 + threadIdx.x;
  if (i < n) p[i] = 0.f;
}

__global__ void axpy_bias_kernel(const float* a, const float* c, float s, float* o) {
  int i = threadIdx.x;  // 256 threads
  o[i] = a[i] + s * c[i];
}

__global__ void mag_kernel(float* __restrict__ sr, const float* __restrict__ si, long n) {
  long i = (long)blockIdx.x * 256 + threadIdx.x;
  if (i < n) sr[i] = sqrtf(sr[i] * sr[i] + si[i] * si[i]);
}

__global__ void buildc_kernel(const float* __restrict__ re, const float* __restrict__ im,
                              float2* __restrict__ oc, long n) {
  long i = (long)blockIdx.x * 256 + threadIdx.x;
  if (i < n) oc[i] = make_float2(re[i], im[i]);
}

// (b,n,d) complex -> mem (n,b,d) real, plus bf16 copy
__global__ void mem_kernel(const float2* __restrict__ oc, float* __restrict__ memf,
                           __bf16* __restrict__ membf) {
  int i = blockIdx.x * 256 + threadIdx.x;
  if (i >= 1048576) return;
  int d = i & 255, n = (i >> 8) & 1023, b = i >> 18;
  float v = oc[i].x;
  long o = (long)n * 1024 + b * 256 + d;
  memf[o] = v; membf[o] = (__bf16)v;
}

__global__ void predsT_kernel(const float* __restrict__ t, float* __restrict__ out) {
  int i = blockIdx.x * 256 + threadIdx.x;
  if (i >= 524288) return;
  int n = i & 1023, c = (i >> 10) & 127, b = i >> 17;
  out[i] = t[((size_t)b * 1024 + n) * 128 + c];
}

// ---------------------------------------------------------------------------
// iwca gate path
// ---------------------------------------------------------------------------
#define BN_SCALE 0.9999950000374997f  // 1/sqrt(1+1e-5)

__global__ void iwca_c0_kernel(const float* __restrict__ x, const float* __restrict__ wgt,
                               const float* __restrict__ bias, const float* __restrict__ g,
                               const float* __restrict__ bb, float* __restrict__ y) {
  int i = blockIdx.x * 256 + threadIdx.x;
  if (i >= 4 * 128 * 64) return;
  int w = i & 7, h = (i >> 3) & 7, c = (i >> 6) & 127, b = i >> 13;
  const float* src = x + ((size_t)(b * 128 + c) << 6);
  float acc = bias[c];
  for (int dy = 0; dy < 3; dy++) {
    int hh = h + dy - 1; if (hh < 0 || hh >= 8) continue;
    for (int dx = 0; dx < 3; dx++) {
      int ww = w + dx - 1; if (ww < 0 || ww >= 8) continue;
      acc += src[hh * 8 + ww] * wgt[c * 9 + dy * 3 + dx];
    }
  }
  acc = fmaxf(acc, 0.f);
  y[i] = acc * (g[c] * BN_SCALE) + bb[c];
}

__global__ void iwca_c1_kernel(const float* __restrict__ x, const float* __restrict__ wgt,
                               const float* __restrict__ bias, const float* __restrict__ g,
                               const float* __restrict__ bb, float* __restrict__ y) {
  int i = blockIdx.x * 256 + threadIdx.x;
  if (i >= 4 * 128 * 64) return;
  int c = (i >> 6) & 127;
  float acc = fmaxf(x[i] * wgt[c] + bias[c], 0.f);
  y[i] = acc * (g[c] * BN_SCALE) + bb[c];
}

__global__ __launch_bounds__(64) void gate_kernel(const float* __restrict__ x, float* __restrict__ gate) {
  __shared__ float red[64];
  int bc = blockIdx.x, tid = threadIdx.x;
  red[tid] = x[bc * 64 + tid];
  __syncthreads();
  for (int k = 32; k > 0; k >>= 1) { if (tid < k) red[tid] += red[tid + k]; __syncthreads(); }
  if (tid == 0) gate[bc] = 1.f / (1.f + __expf(-red[0] / 64.f));
}

__global__ void gatemul_kernel(const float* __restrict__ x, const float* __restrict__ gate,
                               float* __restrict__ y) {
  int i = blockIdx.x * 256 + threadIdx.x;
  if (i < 4 * 128 * 64) y[i] = x[i] * gate[i >> 6];
}

// ---------------------------------------------------------------------------
// plain convs (tiny -> VALU)
// ---------------------------------------------------------------------------
__global__ void conv1x1_relu_kernel(const float* __restrict__ x, const float* __restrict__ wgt,
                                    const float* __restrict__ bias, float* __restrict__ y,
                                    int Cin, int Cout, int HW) {
  int i = blockIdx.x * 256 + threadIdx.x;
  int total = 4 * Cout * HW;
  if (i >= total) return;
  int p = i % HW, co = (i / HW) % Cout, b = i / (HW * Cout);
  const float* src = x + (size_t)b * Cin * HW + p;
  const float* wr  = wgt + (size_t)co * Cin;
  float acc = bias[co];
  for (int ci = 0; ci < Cin; ci++) acc += src[(size_t)ci * HW] * wr[ci];
  y[i] = fmaxf(acc, 0.f);
}

__global__ void conv3x3_relu_kernel(const float* __restrict__ x, const float* __restrict__ wgt,
                                    const float* __restrict__ bias, float* __restrict__ y,
                                    int C, int H) {
  int i = blockIdx.x * 256 + threadIdx.x;
  int total = 4 * C * H * H;
  if (i >= total) return;
  int w = i % H, h = (i / H) % H, co = (i / (H * H)) % C, b = i / (H * H * C);
  float acc = bias[co];
  for (int ci = 0; ci < C; ci++) {
    const float* src = x + (size_t)(b * C + ci) * H * H;
    const float* wr  = wgt + (size_t)(co * C + ci) * 9;
    for (int dy = 0; dy < 3; dy++) {
      int hh = h + dy - 1; if (hh < 0 || hh >= H) continue;
      for (int dx = 0; dx < 3; dx++) {
        int ww = w + dx - 1; if (ww < 0 || ww >= H) continue;
        acc += src[hh * H + ww] * wr[dy * 3 + dx];
      }
    }
  }
  y[i] = fmaxf(acc, 0.f);
}

// upsample z_hsi 8x8 -> 32x32, concat with z_msi, also copy tuple outputs 2,3
__global__ void concat_kernel(const float* __restrict__ zh, const float* __restrict__ zm,
                              float* __restrict__ z, float* __restrict__ out1,
                              float* __restrict__ out2) {
  int i = blockIdx.x * 256 + threadIdx.x;
  if (i >= 1048576) return;
  int w = i & 31, h = (i >> 5) & 31, c = (i >> 10) & 255, b = i >> 18;
  float v;
  if (c < 128) {
    v = zh[((b * 128 + c) << 6) + (h >> 2) * 8 + (w >> 2)];
    out1[(((size_t)b * 128 + c) << 10) + h * 32 + w] = v;
  } else {
    v = zm[(((size_t)b * 128 + (c - 128)) << 10) + h * 32 + w];
    out2[(((size_t)b * 128 + (c - 128)) << 10) + h * 32 + w] = v;
  }
  z[i] = v;
}

// ---------------------------------------------------------------------------
// FFT kernels (radix-2 Cooley-Tukey in LDS)
// ---------------------------------------------------------------------------
__device__ __forceinline__ float2 cmulf(float2 a, float2 b) {
  return make_float2(a.x * b.x - a.y * b.y, a.x * b.y + a.y * b.x);
}
#define PI_F 3.14159265358979323846f

// forward fft2 of one 32x32 real tile per block; outputs bf16 xr, xi (flat layout)
__global__ __launch_bounds__(256)
void fft2_32_kernel(const float* __restrict__ z, __bf16* __restrict__ xr, __bf16* __restrict__ xi) {
  __shared__ float2 t[1024];
  const int bc = blockIdx.x;                 // b*256 + c
  const float* src = z + (size_t)bc * 1024;
  const int tid = threadIdx.x;
  for (int i = tid; i < 1024; i += 256) t[i] = make_float2(src[i], 0.f);
  __syncthreads();
  // ---- FFT along w (rows) ----
  for (int i = tid; i < 1024; i += 256) {
    int h = i >> 5, w = i & 31, r = __brev((unsigned)w) >> 27;
    if (w < r) { float2 a = t[h * 32 + w]; t[h * 32 + w] = t[h * 32 + r]; t[h * 32 + r] = a; }
  }
  __syncthreads();
  for (int s = 1; s <= 5; ++s) {
    int m = 1 << s, mh = m >> 1;
    for (int j = tid; j < 512; j += 256) {
      int h = j >> 4, bj = j & 15;
      int blk = bj / mh, twi = bj % mh;
      int i0 = h * 32 + blk * m + twi, i1 = i0 + mh;
      float ang = -2.f * PI_F * twi / m;
      float2 wv = make_float2(__cosf(ang), __sinf(ang));
      float2 u = t[i0], v = cmulf(t[i1], wv);
      t[i0] = make_float2(u.x + v.x, u.y + v.y);
      t[i1] = make_float2(u.x - v.x, u.y - v.y);
    }
    __syncthreads();
  }
  // ---- FFT along h (cols) ----
  for (int i = tid; i < 1024; i += 256) {
    int h = i >> 5, w = i & 31, r = __brev((unsigned)h) >> 27;
    if (h < r) { float2 a = t[h * 32 + w]; t[h * 32 + w] = t[r * 32 + w]; t[r * 32 + w] = a; }
  }
  __syncthreads();
  for (int s = 1; s <= 5; ++s) {
    int m = 1 << s, mh = m >> 1;
    for (int j = tid; j < 512; j += 256) {
      int w = j & 31, bj = j >> 5;
      int blk = bj / mh, twi = bj % mh;
      int i0 = (blk * m + twi) * 32 + w, i1 = i0 + mh * 32;
      float ang = -2.f * PI_F * twi / m;
      float2 wv = make_float2(__cosf(ang), __sinf(ang));
      float2 u = t[i0], v = cmulf(t[i1], wv);
      t[i0] = make_float2(u.x + v.x, u.y + v.y);
      t[i1] = make_float2(u.x - v.x, u.y - v.y);
    }
    __syncthreads();
  }
  for (int i = tid; i < 1024; i += 256) {
    xr[(size_t)bc * 1024 + i] = (__bf16)t[i].x;
    xi[(size_t)bc * 1024 + i] = (__bf16)t[i].y;
  }
}

// in-place inverse FFT of length L along a strided axis.
// signal base = (bid/per)*chunkA + (bid%per)*chunkB, element n at base + n*stride
__global__ void ifft_kernel(float2* __restrict__ data, int L, int log2L,
                            long chunkA, int per, long chunkB, long stride, float scale) {
  __shared__ float2 t[1024];
  int bid = blockIdx.x, tid = threadIdx.x, nt = blockDim.x;
  long base = (long)(bid / per) * chunkA + (long)(bid % per) * chunkB;
  for (int i = tid; i < L; i += nt) t[i] = data[base + (long)i * stride];
  __syncthreads();
  for (int i = tid; i < L; i += nt) {
    int r = (int)(__brev((unsigned)i) >> (32 - log2L));
    if (i < r) { float2 a = t[i]; t[i] = t[r]; t[r] = a; }
  }
  __syncthreads();
  for (int s = 1; s <= log2L; ++s) {
    int m = 1 << s, mh = m >> 1;
    for (int j = tid; j < (L >> 1); j += nt) {
      int blk = j / mh, twi = j % mh;
      int i0 = blk * m + twi, i1 = i0 + mh;
      float ang = 2.f * PI_F * twi / m;  // +i for inverse
      float2 wv = make_float2(__cosf(ang), __sinf(ang));
      float2 u = t[i0], v = cmulf(t[i1], wv);
      t[i0] = make_float2(u.x + v.x, u.y + v.y);
      t[i1] = make_float2(u.x - v.x, u.y - v.y);
    }
    __syncthreads();
  }
  for (int i = tid; i < L; i += nt)
    data[base + (long)i * stride] = make_float2(t[i].x * scale, t[i].y * scale);
}

// ---------------------------------------------------------------------------
// softmax over rows of length L (in place f32, optional bf16 copy)
// ---------------------------------------------------------------------------
__global__ __launch_bounds__(256)
void softmax_kernel(float* __restrict__ x, __bf16* __restrict__ xb, int L) {
  __shared__ float red[256];
  long row = blockIdx.x;
  float* p = x + row * (long)L;
  int tid = threadIdx.x;
  float mx = -1e30f;
  for (int i = tid; i < L; i += 256) mx = fmaxf(mx, p[i]);
  red[tid] = mx; __syncthreads();
  for (int s = 128; s > 0; s >>= 1) { if (tid < s) red[tid] = fmaxf(red[tid], red[tid + s]); __syncthreads(); }
  mx = red[0]; __syncthreads();
  float sum = 0.f;
  for (int i = tid; i < L; i += 256) { float e = __expf(p[i] - mx); p[i] = e; sum += e; }
  red[tid] = sum; __syncthreads();
  for (int s = 128; s > 0; s >>= 1) { if (tid < s) red[tid] += red[tid + s]; __syncthreads(); }
  float inv = 1.f / red[0];
  for (int i = tid; i < L; i += 256) {
    float v = p[i] * inv; p[i] = v;
    if (xb) xb[row * (long)L + i] = (__bf16)v;
  }
}

// ---------------------------------------------------------------------------
// layernorm over 256 features: y = LN(x + res) * g + b   (in-place safe)
// ---------------------------------------------------------------------------
__global__ __launch_bounds__(64)
void ln_kernel(const float* __restrict__ x, const float* __restrict__ r,
               const float* __restrict__ g, const float* __restrict__ b,
               float* __restrict__ y) {
  __shared__ float red[64];
  __shared__ float buf[256];
  long row = blockIdx.x;
  int tid = threadIdx.x;
  const float* px = x + row * 256;
  const float* pr = r ? r + row * 256 : nullptr;
  float s = 0.f;
  for (int i = tid; i < 256; i += 64) { float t = px[i] + (pr ? pr[i] : 0.f); buf[i] = t; s += t; }
  red[tid] = s; __syncthreads();
  for (int k = 32; k > 0; k >>= 1) { if (tid < k) red[tid] += red[tid + k]; __syncthreads(); }
  float mean = red[0] / 256.f; __syncthreads();
  s = 0.f;
  for (int i = tid; i < 256; i += 64) { float d = buf[i] - mean; s += d * d; }
  red[tid] = s; __syncthreads();
  for (int k = 32; k > 0; k >>= 1) { if (tid < k) red[tid] += red[tid + k]; __syncthreads(); }
  float inv = rsqrtf(red[0] / 256.f + 1e-5f);
  for (int i = tid; i < 256; i += 64) y[row * 256 + i] = (buf[i] - mean) * inv * g[i] + b[i];
}

// ---------------------------------------------------------------------------
// host orchestration
// ---------------------------------------------------------------------------
static inline unsigned gblk(long n) { return (unsigned)((n + 255) / 256); }

extern "C" void kernel_launch(void* const* d_in, const int* in_sizes, int n_in,
                              void* d_out, int out_size, void* d_ws, size_t ws_size,
                              hipStream_t stream) {
  (void)in_sizes; (void)n_in; (void)out_size; (void)ws_size;

  const float* hsi = (const float*)d_in[0];
  const float* msi = (const float*)d_in[1];
  int pi = 2;
  auto P = [&]() -> const float* { return (const float*)d_in[pi++]; };

  // params in setup_inputs() insertion order (depth-first)
  const float *c0w = P(), *c0b = P(), *bn0g = P(), *bn0b = P();
  const float *c1w = P(), *c1b = P(), *bn1g = P(), *bn1b = P();
  struct CB { const float *w1, *b1, *w2, *b2, *w3, *b3; } he[3], me[3];
  for (int i = 0; i < 3; i++) { he[i].w1 = P(); he[i].b1 = P(); he[i].w2 = P(); he[i].b2 = P(); he[i].w3 = P(); he[i].b3 = P(); }
  for (int i = 0; i < 3; i++) { me[i].w1 = P(); me[i].b1 = P(); me[i].w2 = P(); me[i].b2 = P(); me[i].w3 = P(); me[i].b3 = P(); }
  const float *fw[6], *fb[6];                       // qr, qi, kr, ki, vr, vi
  for (int j = 0; j < 6; j++) { fw[j] = P(); fb[j] = P(); }
  struct DL { const float *saw, *sab, *sow, *sob, *caw, *cab, *cow, *cob,
                          *f1w, *f1b, *f2w, *f2b, *l1g, *l1b, *l2g, *l2b, *l3g, *l3b; } Ld[6];
  for (int l = 0; l < 6; l++) {
    Ld[l].saw = P(); Ld[l].sab = P(); Ld[l].sow = P(); Ld[l].sob = P();
    Ld[l].caw = P(); Ld[l].cab = P(); Ld[l].cow = P(); Ld[l].cob = P();
    Ld[l].f1w = P(); Ld[l].f1b = P(); Ld[l].f2w = P(); Ld[l].f2b = P();
    Ld[l].l1g = P(); Ld[l].l1b = P(); Ld[l].l2g = P(); Ld[l].l2b = P();
    Ld[l].l3g = P(); Ld[l].l3b = P();
  }
  const float *linw = P(), *linb = P();

  // ---- workspace bump allocator ----
  char* wsb = (char*)d_ws;
  size_t off = 0;
  auto alloc = [&](size_t bytes) -> void* {
    off = (off + 255) & ~(size_t)255;
    void* p = wsb + off; off += bytes; return p;
  };

  // shared (whole-launch) buffers
  float*  z      = (float*)alloc(1048576 * 4);
  float*  t0     = (float*)alloc(524288 * 4);
  float*  t1     = (float*)alloc(524288 * 4);
  float*  zh     = (float*)alloc(32768 * 4);
  float*  gate   = (float*)alloc(512 * 4);
  float*  memf   = (float*)alloc(1048576 * 4);
  __bf16* membf  = (__bf16*)alloc(1048576 * 2);
  float*  xbuf   = (float*)alloc(1048576 * 4);
  __bf16* xbf    = (__bf16*)alloc(1048576 * 2);
  float*  predt  = (float*)alloc(524288 * 4);
  float*  bcomb  = (float*)alloc(6 * 256 * 4);     // bqm,bqp,bkm,bkp,bvm,bvp
  __bf16* fcawbf[6];
  for (int j = 0; j < 6; j++) fcawbf[j] = (__bf16*)alloc(65536 * 2);
  __bf16 *saInBf[6], *saOutBf[6], *caInBf[6], *caOutBf[6], *ff1Bf[6], *ff2Bf[6];
  for (int l = 0; l < 6; l++) {
    saInBf[l]  = (__bf16*)alloc(196608 * 2);
    saOutBf[l] = (__bf16*)alloc(65536 * 2);
    caInBf[l]  = (__bf16*)alloc(196608 * 2);
    caOutBf[l] = (__bf16*)alloc(65536 * 2);
    ff1Bf[l]   = (__bf16*)alloc(524288 * 2);
    ff2Bf[l]   = (__bf16*)alloc(524288 * 2);
  }
  __bf16* linbf = (__bf16*)alloc(32768 * 2);
  size_t scratch = off;

  // phase 1: frequency cross-attention scratch
  __bf16* xrbf = (__bf16*)alloc(1048576 * 2);
  __bf16* xibf = (__bf16*)alloc(1048576 * 2);
  float *Qr = (float*)alloc(1048576 * 4), *Qi = (float*)alloc(1048576 * 4);
  float *Kr = (float*)alloc(1048576 * 4), *Ki = (float*)alloc(1048576 * 4);
  float *Vr = (float*)alloc(1048576 * 4), *Vi = (float*)alloc(1048576 * 4);
  __bf16 *Qrb = (__bf16*)alloc(1048576 * 2), *Qib = (__bf16*)alloc(1048576 * 2);
  __bf16 *Krb = (__bf16*)alloc(1048576 * 2), *Kib = (__bf16*)alloc(1048576 * 2);
  __bf16 *Vrb = (__bf16*)alloc(1048576 * 2), *Vib = (__bf16*)alloc(1048576 * 2);
  float*  sr    = (float*)alloc(4194304 * 4);
  float*  si    = (float*)alloc(4194304 * 4);
  __bf16* attbf = (__bf16*)alloc(4194304 * 2);
  float*  outR  = (float*)alloc(1048576 * 4);
  float*  outI  = (float*)alloc(1048576 * 4);
  float2* oc    = (float2*)alloc(1048576 * 8);

  // phase 2: decoder scratch (reuses phase-1 region; stream order keeps it safe)
  off = scratch;
  float*  qkv    = (float*)alloc(3145728 * 4);
  __bf16* qkvbf  = (__bf16*)alloc(3145728 * 2);
  float*  scores = (float*)alloc(8388608 * 4);
  __bf16* probbf = (__bf16*)alloc(8388608 * 2);
  float*  obuf   = (float*)alloc(1048576 * 4);
  __bf16* obf    = (__bf16*)alloc(1048576 * 2);
  float*  attno  = (float*)alloc(1048576 * 4);
  float*  ff1    = (float*)alloc(8388608 * 4);
  __bf16* ff1bf  = (__bf16*)alloc(8388608 * 2);

  float* outPred = (float*)d_out;
  float* out1 = outPred + 524288;   // z_hsi_up
  float* out2 = outPred + 1048576;  // z_msi

  auto pack = [&](const float* s, __bf16* dptr, long n) {
    pack_bf16_kernel<<<gblk(n), 256, 0, stream>>>(s, dptr, n);
  };
  // dispatch: sBn==1 -> B staged n-contiguous (BT); else sBk==1 (k contiguous)
  auto gemm = [&](const __bf16* A, const __bf16* B, float* C, const float* bias,
                  int M, int N, int K,
                  long sAm, long sAk, long sBk, long sBn, long sCm,
                  long bA, long bB, long bC, int batch, float alpha, int flags) {
    (void)sAk;
    if (sBn == 1) {
      if ((N & 63) == 0) {
        dim3 g(N / 64, M / 64, batch);
        gemm_fast_kernel<64, true><<<g, 128, 0, stream>>>(
            A, B, C, bias, K, sAm, sBk, sBn, sCm, bA, bB, bC, alpha, flags);
      } else {
        dim3 g(N / 32, M / 64, batch);
        gemm_fast_kernel<32, true><<<g, 128, 0, stream>>>(
            A, B, C, bias, K, sAm, sBk, sBn, sCm, bA, bB, bC, alpha, flags);
      }
    } else {
      dim3 g(N / 64, M / 64, batch);
      gemm_fast_kernel<64, false><<<g, 128, 0, stream>>>(
          A, B, C, bias, K, sAm, sBk, sBn, sCm, bA, bB, bC, alpha, flags);
    }
  };

  // ---- pack all weights to bf16 ----
  for (int j = 0; j < 6; j++) pack(fw[j], fcawbf[j], 65536);
  for (int l = 0; l < 6; l++) {
    pack(Ld[l].saw, saInBf[l], 196608);
    pack(Ld[l].sow, saOutBf[l], 65536);
    pack(Ld[l].caw, caInBf[l], 196608);
    pack(Ld[l].cow, caOutBf[l], 65536);
    pack(Ld[l].f1w, ff1Bf[l], 524288);
    pack(Ld[l].f2w, ff2Bf[l], 524288);
  }
  pack(linw, linbf, 32768);
  // combined complex-linear biases
  axpy_bias_kernel<<<1, 256, 0, stream>>>(fb[0], fb[1], -1.f, bcomb + 0);
  axpy_bias_kernel<<<1, 256, 0, stream>>>(fb[0], fb[1], 1.f, bcomb + 256);
  axpy_bias_kernel<<<1, 256, 0, stream>>>(fb[2], fb[3], -1.f, bcomb + 512);
  axpy_bias_kernel<<<1, 256, 0, stream>>>(fb[2], fb[3], 1.f, bcomb + 768);
  axpy_bias_kernel<<<1, 256, 0, stream>>>(fb[4], fb[5], -1.f, bcomb + 1024);
  axpy_bias_kernel<<<1, 256, 0, stream>>>(fb[4], fb[5], 1.f, bcomb + 1280);

  // ---- iwca gate + encoders ----
  iwca_c0_kernel<<<gblk(32768), 256, 0, stream>>>(hsi, c0w, c0b, bn0g, bn0b, t0);
  iwca_c1_kernel<<<gblk(32768), 256, 0, stream>>>(t0, c1w, c1b, bn1g, bn1b, t1);
  gate_kernel<<<512, 64, 0, stream>>>(t1, gate);
  gatemul_kernel<<<gblk(32768), 256, 0, stream>>>(hsi, gate, t0);
  // hsi encoder (8x8)
  conv1x1_relu_kernel<<<gblk(4 * 64 * 64), 256, 0, stream>>>(t0, he[0].w1, he[0].b1, t1, 128, 64, 64);
  conv3x3_relu_kernel<<<gblk(4 * 64 * 64), 256, 0, stream>>>(t1, he[0].w2, he[0].b2, t0, 64, 8);
  conv1x1_relu_kernel<<<gblk(4 * 64 * 64), 256, 0, stream>>>(t0, he[0].w3, he[0].b3, t1, 64, 64, 64);
  conv1x1_relu_kernel<<<gblk(4 * 128 * 64), 256, 0, stream>>>(t1, he[1].w1, he[1].b1, t0, 64, 128, 64);
  conv3x3_relu_kernel<<<gblk(4 * 128 * 64), 256, 0, stream>>>(t0, he[1].w2, he[1].b2, t1, 128, 8);
  conv1x1_relu_kernel<<<gblk(4 * 128 * 64), 256, 0, stream>>>(t1, he[1].w3, he[1].b3, t0, 128, 128, 64);
  conv1x1_relu_kernel<<<gblk(4 * 128 * 64), 256, 0, stream>>>(t0, he[2].w1, he[2].b1, t1, 128, 128, 64);
  conv3x3_relu_kernel<<<gblk(4 * 128 * 64), 256, 0, stream>>>(t1, he[2].w2, he[2].b2, t0, 128, 8);
  conv1x1_relu_kernel<<<gblk(4 * 128 * 64), 256, 0, stream>>>(t0, he[2].w3, he[2].b3, zh, 128, 128, 64);
  // msi encoder (32x32)
  conv1x1_relu_kernel<<<gblk(4 * 64 * 1024), 256, 0, stream>>>(msi, me[0].w1, me[0].b1, t0, 4, 64, 1024);
  conv3x3_relu_kernel<<<gblk(4 * 64 * 1024), 256, 0, stream>>>(t0, me[0].w2, me[0].b2, t1, 64, 32);
  conv1x1_relu_kernel<<<gblk(4 * 64 * 1024), 256, 0, stream>>>(t1, me[0].w3, me[0].b3, t0, 64, 64, 1024);
  conv1x1_relu_kernel<<<gblk(4 * 128 * 1024), 256, 0, stream>>>(t0, me[1].w1, me[1].b1, t1, 64, 128, 1024);
  conv3x3_relu_kernel<<<gblk(4 * 128 * 1024), 256, 0, stream>>>(t1, me[1].w2, me[1].b2, t0, 128, 32);
  conv1x1_relu_kernel<<<gblk(4 * 128 * 1024), 256, 0, stream>>>(t0, me[1].w3, me[1].b3, t1, 128, 128, 1024);
  conv1x1_relu_kernel<<<gblk(4 * 128 * 1024), 256, 0, stream>>>(t1, me[2].w1, me[2].b1, t0, 128, 128, 1024);
  conv3x3_relu_kernel<<<gblk(4 * 128 * 1024), 256, 0, stream>>>(t0, me[2].w2, me[2].b2, t1, 128, 32);
  conv1x1_relu_kernel<<<gblk(4 * 128 * 1024), 256, 0, stream>>>(t1, me[2].w3, me[2].b3, t0, 128, 128, 1024);
  concat_kernel<<<gblk(1048576), 256, 0, stream>>>(zh, t0, z, out1, out2);

  // ---- fft2 + complex linears ----
  fft2_32_kernel<<<1024, 256, 0, stream>>>(z, xrbf, xibf);
  // Qr = xr@Wqr^T - xi@Wqi^T + (bqr-bqi) ; Qi = xi@Wqr^T + xr@Wqi^T + (bqr+bqi)
  gemm(xrbf, fcawbf[0], Qr, bcomb + 0, 4096, 256, 256, 256, 1, 1, 256, 256, 0, 0, 0, 1, 1.f, 0);
  gemm(xibf, fcawbf[1], Qr, nullptr,   4096, 256, 256, 256, 1, 1, 256, 256, 0, 0, 0, 1, -1.f, 1);
  gemm(xibf, fcawbf[0], Qi, bcomb + 256, 4096, 256, 256, 256, 1, 1, 256, 256, 0, 0, 0, 1, 1.f, 0);
  gemm(xrbf, fcawbf[1], Qi, nullptr,     4096, 256, 256, 256, 1, 1, 256, 256, 0, 0, 0, 1, 1.f, 1);
  gemm(xrbf, fcawbf[2], Kr, bcomb + 512, 4096, 256, 256, 256, 1, 1, 256, 256, 0, 0, 0, 1, 1.f, 0);
  gemm(xibf, fcawbf[3], Kr, nullptr,     4096, 256, 256, 256, 1, 1, 256, 256, 0, 0, 0, 1, -1.f, 1);
  gemm(xibf, fcawbf[2], Ki, bcomb + 768, 4096, 256, 256, 256, 1, 1, 256, 256, 0, 0, 0, 1, 1.f, 0);
  gemm(xrbf, fcawbf[3], Ki, nullptr,     4096, 256, 256, 256, 1, 1, 256, 256, 0, 0, 0, 1, 1.f, 1);
  gemm(xrbf, fcawbf[4], Vr, bcomb + 1024, 4096, 256, 256, 256, 1, 1, 256, 256, 0, 0, 0, 1, 1.f, 0);
  gemm(xibf, fcawbf[5], Vr, nullptr,      4096, 256, 256, 256, 1, 1, 256, 256, 0, 0, 0, 1, -1.f, 1);
  gemm(xibf, fcawbf[4], Vi, bcomb + 1280, 4096, 256, 256, 256, 1, 1, 256, 256, 0, 0, 0, 1, 1.f, 0);
  gemm(xrbf, fcawbf[5], Vi, nullptr,      4096, 256, 256, 256, 1, 1, 256, 256, 0, 0, 0, 1, 1.f, 1);
  pack(Qr, Qrb, 1048576); pack(Qi, Qib, 1048576);
  pack(Kr, Krb, 1048576); pack(Ki, Kib, 1048576);
  pack(Vr, Vrb, 1048576); pack(Vi, Vib, 1048576);

  // sr = QrKr^T - QiKi^T ; si = QrKi^T + QiKr^T  (batched over B=4)
  gemm(Qrb, Krb, sr, nullptr, 1024, 1024, 256, 256, 1, 1, 256, 1024, 262144, 262144, 1048576, 4, 1.f, 0);
  gemm(Qib, Kib, sr, nullptr, 1024, 1024, 256, 256, 1, 1, 256, 1024, 262144, 262144, 1048576, 4, -1.f, 1);
  gemm(Qrb, Kib, si, nullptr, 1024, 1024, 256, 256, 1, 1, 256, 1024, 262144, 262144, 1048576, 4, 1.f, 0);
  gemm(Qib, Krb, si, nullptr, 1024, 1024, 256, 256, 1, 1, 256, 1024, 262144, 262144, 1048576, 4, 1.f, 1);
  mag_kernel<<<gblk(4194304), 256, 0, stream>>>(sr, si, 4194304);
  softmax_kernel<<<4096, 256, 0, stream>>>(sr, attbf, 1024);
  gemm(attbf, Vrb, outR, nullptr, 1024, 256, 1024, 1024, 1, 256, 1, 256, 1048576, 262144, 262144, 4, 1.f, 0);
  gemm(attbf, Vib, outI, nullptr, 1024, 256, 1024, 1024, 1, 256, 1, 256, 1048576, 262144, 262144, 4, 1.f, 0);
  buildc_kernel<<<gblk(1048576), 256, 0, stream>>>(outR, outI, oc, 1048576);
  // ifft2 over (1024, 256) per batch
  ifft_kernel<<<1024, 512, 0, stream>>>(oc, 1024, 10, 262144, 256, 1, 256, 1.f / 1024.f);
  ifft_kernel<<<4096, 128, 0, stream>>>(oc, 256, 8, 256, 1, 0, 1, 1.f / 256.f);
  mem_kernel<<<gblk(1048576), 256, 0, stream>>>(oc, memf, membf);

  // ---- transformer decoder ----
  zero_kernel<<<gblk(1048576), 256, 0, stream>>>(xbuf, 1048576);

  auto attn_core = [&](const __bf16* qkvb) {
    for (int b = 0; b < 4; b++) {
      gemm(qkvb + (size_t)b * 768, qkvb + (size_t)b * 768 + 256, scores, nullptr,
           1024, 1024, 32, 3072, 1, 1, 3072, 1024, 32, 32, 1048576, 8,
           0.17677669529663689f, 0);
      softmax_kernel<<<8192, 256, 0, stream>>>(scores, probbf, 1024);
      gemm(probbf, qkvb + (size_t)b * 768 + 512, obuf + (size_t)b * 256, nullptr,
           1024, 32, 1024, 1024, 1, 3072, 1, 1024, 1048576, 32, 32, 8, 1.f, 0);
    }
  };

  for (int l = 0; l < 6; l++) {
    const DL& dl = Ld[l];
    // self attention
    pack(xbuf, xbf, 1048576);
    gemm(xbf, saInBf[l], qkv, dl.sab, 4096, 768, 256, 256, 1, 1, 256, 768, 0, 0, 0, 1, 1.f, 0);
    pack(qkv, qkvbf, 3145728);
    attn_core(qkvbf);
    pack(obuf, obf, 1048576);
    gemm(obf, saOutBf[l], attno, dl.sob, 4096, 256, 256, 256, 1, 1, 256, 256, 0, 0, 0, 1, 1.f, 0);
    ln_kernel<<<4096, 64, 0, stream>>>(xbuf, attno, dl.l1g, dl.l1b, xbuf);
    // cross attention (q from x, k/v from mem)
    pack(xbuf, xbf, 1048576);
    gemm(xbf, caInBf[l], qkv, dl.cab, 4096, 256, 256, 256, 1, 1, 256, 768, 0, 0, 0, 1, 1.f, 0);
    gemm(membf, caInBf[l] + 65536, qkv + 256, dl.cab + 256, 4096, 512, 256,
         256, 1, 1, 256, 768, 0, 0, 0, 1, 1.f, 0);
    pack(qkv, qkvbf, 3145728);
    attn_core(qkvbf);
    pack(obuf, obf, 1048576);
    gemm(obf, caOutBf[l], attno, dl.cob, 4096, 256, 256, 256, 1, 1, 256, 256, 0, 0, 0, 1, 1.f, 0);
    ln_kernel<<<4096, 64, 0, stream>>>(xbuf, attno, dl.l2g, dl.l2b, xbuf);
    // feed-forward
    pack(xbuf, xbf, 1048576);
    gemm(xbf, ff1Bf[l], ff1, dl.f1b, 4096, 2048, 256, 256, 1, 1, 256, 2048, 0, 0, 0, 1, 1.f, 2);
    pack(ff1, ff1bf, 8388608);
    gemm(ff1bf, ff2Bf[l], attno, dl.f2b, 4096, 256, 2048, 2048, 1, 1, 2048, 256, 0, 0, 0, 1, 1.f, 0);
    ln_kernel<<<4096, 64, 0, stream>>>(xbuf, attno, dl.l3g, dl.l3b, xbuf);
  }

  // ---- output head ----
  pack(xbuf, xbf, 1048576);
  gemm(xbf, linbf, predt, linb, 1024, 128, 256, 1024, 1, 1, 256, 128,
       256, 0, 131072, 4, 1.f, 0);
  predsT_kernel<<<gblk(524288), 256, 0, stream>>>(predt, outPred);
}
